// ClinicalGCN_70858370450169
// MI455X (gfx1250) — compile-verified, matrix operand order checked by reading, not run on hardware
//
#include <hip/hip_runtime.h>

#define NNODES   100000
#define NEDGES   1600000
#define HDIM     128
#define GGRAPHS  256
#define CCLIN    16
#define KOUT     2
#define EPSBN    1e-5f

typedef __attribute__((ext_vector_type(16))) _Float16 v16h;
typedef __attribute__((ext_vector_type(8)))  float    v8f;

// ---------------------------------------------------------------------------
// degree: dis[] temporarily holds degree counts
__global__ void deg_kernel(const int* __restrict__ ei, float* __restrict__ deg) {
    long long e = (long long)blockIdx.x * blockDim.x + threadIdx.x;
    if (e >= NEDGES) return;
    int dst = ei[NEDGES + e];
    atomicAdd(&deg[dst], 1.0f);
}

__global__ void rsqrt_kernel(float* __restrict__ dis) {
    long long n = (long long)blockIdx.x * blockDim.x + threadIdx.x;
    if (n >= NNODES) return;
    dis[n] = rsqrtf(dis[n] + 1.0f);
}

__global__ void init_affine_kernel(float* __restrict__ scale, float* __restrict__ shift) {
    int f = threadIdx.x;
    scale[f] = 1.0f;
    shift[f] = 0.0f;
}

// ---------------------------------------------------------------------------
// WMMA GEMM: xw[n][j] = sum_k (hin[n][k]*scale[k]+shift[k]) * W[k][j]
// One wave -> 16 rows x 128 cols.  8 waves / block -> 128 rows / block.
__global__ __launch_bounds__(256) void gemm_wmma_kernel(
    const float* __restrict__ hin, const float* __restrict__ scale,
    const float* __restrict__ shift, const float* __restrict__ W,
    float* __restrict__ xw, int nrows)
{
    // W transposed to f16 in LDS, padded row stride 136 (272B = 17*16B: aligned,
    // and 68 dwords ≡ 4 mod 64 banks -> no systematic conflicts)
    __shared__ _Float16 WT[HDIM][HDIM + 8];
    __shared__ float sc[HDIM], sh[HDIM];

    int t = threadIdx.x;
    for (int i = t; i < HDIM * HDIM; i += 256) {
        int k = i >> 7, n = i & (HDIM - 1);
        WT[n][k] = (_Float16)W[i];
    }
    if (t < HDIM) { sc[t] = scale[t]; sh[t] = shift[t]; }
    __syncthreads();

    int wave = t >> 5;
    int lane = t & 31;
    int m0   = (blockIdx.x * 8 + wave) * 16;
    if (m0 >= nrows) return;

    int half = lane >> 4;        // 0: K {0..7,16..23}   1: K {8..15,24..31}
    int r    = lane & 15;        // A: row within tile / B: col within tile
    int koff = half * 8;
    const float* arow = hin + (size_t)(m0 + r) * HDIM;

    v8f acc[8];
#pragma unroll
    for (int nt = 0; nt < 8; ++nt) acc[nt] = (v8f){0,0,0,0,0,0,0,0};

    for (int kk = 0; kk < HDIM; kk += 32) {
        v16h a;
#pragma unroll
        for (int j = 0; j < 8; ++j) {
            int k1 = kk + koff + j;
            int k2 = k1 + 16;
            a[j]     = (_Float16)(arow[k1] * sc[k1] + sh[k1]);
            a[8 + j] = (_Float16)(arow[k2] * sc[k2] + sh[k2]);
        }
#pragma unroll
        for (int nt = 0; nt < 8; ++nt) {
            int n = nt * 16 + r;
            v16h b;
#pragma unroll
            for (int j = 0; j < 8; ++j) {
                b[j]     = WT[n][kk + koff + j];
                b[8 + j] = WT[n][kk + koff + 16 + j];
            }
            acc[nt] = __builtin_amdgcn_wmma_f32_16x16x32_f16(
                false, a, false, b, (short)0, acc[nt], false, false);
        }
    }

    // C/D layout: lane picks column (N = nt*16 + r), vgpr v picks row (M = v + 8*half)
#pragma unroll
    for (int nt = 0; nt < 8; ++nt) {
#pragma unroll
        for (int v = 0; v < 8; ++v) {
            xw[(size_t)(m0 + v + 8 * half) * HDIM + nt * 16 + r] = acc[nt][v];
        }
    }
}

// ---------------------------------------------------------------------------
// edge scatter: agg[dst] += dis[src]*dis[dst] * xw[src]   (32 threads/edge, 4 feats each)
__global__ __launch_bounds__(256) void agg_kernel(
    const int* __restrict__ ei, const float* __restrict__ dis,
    const float* __restrict__ xw, float* __restrict__ agg)
{
    long long gid = (long long)blockIdx.x * blockDim.x + threadIdx.x;
    long long e   = gid >> 5;
    if (e >= NEDGES) return;
    int fo  = (int)(gid & 31) * 4;
    int src = ei[e];
    int dst = ei[NEDGES + e];
    float w = dis[src] * dis[dst];
    float4 v = *(const float4*)(xw + (size_t)src * HDIM + fo);
    float* ad = agg + (size_t)dst * HDIM + fo;
    atomicAdd(ad + 0, w * v.x);
    atomicAdd(ad + 1, w * v.y);
    atomicAdd(ad + 2, w * v.z);
    atomicAdd(ad + 3, w * v.w);
}

// ---------------------------------------------------------------------------
// h = relu(agg + self_norm*xw + bias); accumulate per-feature sum / sumsq
__global__ __launch_bounds__(256) void finalize_kernel(
    const float* __restrict__ agg, const float* __restrict__ xw,
    const float* __restrict__ dis, const float* __restrict__ bias,
    float* __restrict__ hout, float* __restrict__ stats, int nrows)
{
    __shared__ float s0[HDIM], s1[HDIM];
    int t  = threadIdx.x;
    int f  = t & (HDIM - 1);
    int rl = t >> 7;                      // 0 or 1
    int row0 = blockIdx.x * 64;
    int rend = row0 + 64; if (rend > nrows) rend = nrows;
    float bf = bias[f];
    float s = 0.0f, ss = 0.0f;
    for (int r = row0 + rl; r < rend; r += 2) {
        size_t idx = (size_t)r * HDIM + f;
        float sn = dis[r];
        float v  = agg[idx] + sn * sn * xw[idx] + bf;
        v = fmaxf(v, 0.0f);
        hout[idx] = v;
        s += v; ss += v * v;
    }
    if (rl == 1) { s0[f] = s; s1[f] = ss; }
    __syncthreads();
    if (rl == 0) {
        atomicAdd(&stats[f],        s  + s0[f]);
        atomicAdd(&stats[HDIM + f], ss + s1[f]);
    }
}

__global__ void bn_params_kernel(const float* __restrict__ stats,
                                 const float* __restrict__ gamma,
                                 const float* __restrict__ beta,
                                 float* __restrict__ scale, float* __restrict__ shift)
{
    int f = threadIdx.x;
    const float inv_n = 1.0f / (float)NNODES;
    float mean = stats[f] * inv_n;
    float var  = stats[HDIM + f] * inv_n - mean * mean;
    float s    = gamma[f] * rsqrtf(var + EPSBN);
    scale[f] = s;
    shift[f] = beta[f] - mean * s;
}

// ---------------------------------------------------------------------------
// mean-pool accumulation (BN3 affine applied on the fly)
__global__ __launch_bounds__(256) void pool_kernel(
    const float* __restrict__ h, const float* __restrict__ scale,
    const float* __restrict__ shift, const int* __restrict__ batch,
    float* __restrict__ pooled, float* __restrict__ cnt)
{
    long long gid = (long long)blockIdx.x * blockDim.x + threadIdx.x;
    long long n   = gid >> 5;
    if (n >= NNODES) return;
    int fo = (int)(gid & 31) * 4;
    int g  = batch[n];
    if (fo == 0) atomicAdd(&cnt[g], 1.0f);
    float4 v = *(const float4*)(h + (size_t)n * HDIM + fo);
    float* p = pooled + (size_t)g * HDIM + fo;
    atomicAdd(p + 0, v.x * scale[fo + 0] + shift[fo + 0]);
    atomicAdd(p + 1, v.y * scale[fo + 1] + shift[fo + 1]);
    atomicAdd(p + 2, v.z * scale[fo + 2] + shift[fo + 2]);
    atomicAdd(p + 3, v.w * scale[fo + 3] + shift[fo + 3]);
}

// ---------------------------------------------------------------------------
// head: out[g][k] = bc[k] + sum_f pooled[g][f]/cnt[g] * Wc[f][k] + sum_c clin[g][c]*Wc[H+c][k]
__global__ __launch_bounds__(128) void head_kernel(
    const float* __restrict__ pooled, const float* __restrict__ cnt,
    const float* __restrict__ clinical, const float* __restrict__ Wc,
    const float* __restrict__ bc, float* __restrict__ out)
{
    __shared__ float red0[HDIM], red1[HDIM];
    int g = blockIdx.x;
    int t = threadIdx.x;
    float c  = fmaxf(cnt[g], 1.0f);
    float p  = pooled[(size_t)g * HDIM + t] / c;
    float s0 = p * Wc[t * KOUT + 0];
    float s1 = p * Wc[t * KOUT + 1];
    if (t < CCLIN) {
        float cl = clinical[g * CCLIN + t];
        s0 += cl * Wc[(HDIM + t) * KOUT + 0];
        s1 += cl * Wc[(HDIM + t) * KOUT + 1];
    }
    red0[t] = s0; red1[t] = s1;
    __syncthreads();
    for (int off = 64; off > 0; off >>= 1) {
        if (t < off) { red0[t] += red0[t + off]; red1[t] += red1[t + off]; }
        __syncthreads();
    }
    if (t == 0) {
        out[g * KOUT + 0] = red0[0] + bc[0];
        out[g * KOUT + 1] = red1[0] + bc[1];
    }
}

// ---------------------------------------------------------------------------
extern "C" void kernel_launch(void* const* d_in, const int* in_sizes, int n_in,
                              void* d_out, int out_size, void* d_ws, size_t ws_size,
                              hipStream_t stream)
{
    const float* x        = (const float*)d_in[0];
    const int*   ei       = (const int*)  d_in[1];   // [2, E]
    const int*   batch    = (const int*)  d_in[2];   // [N]
    const float* clinical = (const float*)d_in[3];
    const float* Ws[3]  = { (const float*)d_in[4], (const float*)d_in[6], (const float*)d_in[8] };
    const float* bs[3]  = { (const float*)d_in[5], (const float*)d_in[7], (const float*)d_in[9] };
    const float* gs[3]  = { (const float*)d_in[10], (const float*)d_in[12], (const float*)d_in[14] };
    const float* bes[3] = { (const float*)d_in[11], (const float*)d_in[13], (const float*)d_in[15] };
    const float* Wc = (const float*)d_in[16];
    const float* bc = (const float*)d_in[17];
    float* out = (float*)d_out;

    // workspace carve-up (256B aligned)
    char* ws = (char*)d_ws;
    size_t off = 0;
    auto carve = [&](size_t bytes) -> float* {
        float* p = (float*)(ws + off);
        off += (bytes + 255) & ~(size_t)255;
        return p;
    };
    const size_t NH = (size_t)NNODES * HDIM;
    float* dis    = carve((size_t)NNODES * 4);
    float* xw     = carve(NH * 4);
    float* agg    = carve(NH * 4);
    float* hbuf   = carve(NH * 4);
    float* stats  = carve(2 * HDIM * 4);
    float* scale  = carve(HDIM * 4);
    float* shift  = carve(HDIM * 4);
    float* pooled = carve((size_t)GGRAPHS * HDIM * 4);
    float* cntbuf = carve(GGRAPHS * 4);
    (void)ws_size; (void)in_sizes; (void)n_in; (void)out_size;

    // degree -> dis = rsqrt(deg+1)
    hipMemsetAsync(dis, 0, (size_t)NNODES * 4, stream);
    deg_kernel  <<<(NEDGES + 255) / 256, 256, 0, stream>>>(ei, dis);
    rsqrt_kernel<<<(NNODES + 255) / 256, 256, 0, stream>>>(dis);
    init_affine_kernel<<<1, HDIM, 0, stream>>>(scale, shift);

    const long long aggThreads  = (long long)NEDGES * 32;
    const long long poolThreads = (long long)NNODES * 32;

    for (int l = 0; l < 3; ++l) {
        hipMemsetAsync(agg,   0, NH * 4,       stream);
        hipMemsetAsync(stats, 0, 2 * HDIM * 4, stream);
        const float* hin = (l == 0) ? x : hbuf;
        gemm_wmma_kernel<<<(NNODES + 127) / 128, 256, 0, stream>>>(
            hin, scale, shift, Ws[l], xw, NNODES);
        agg_kernel<<<(unsigned)((aggThreads + 255) / 256), 256, 0, stream>>>(
            ei, dis, xw, agg);
        finalize_kernel<<<(NNODES + 63) / 64, 256, 0, stream>>>(
            agg, xw, dis, bs[l], hbuf, stats, NNODES);
        bn_params_kernel<<<1, HDIM, 0, stream>>>(stats, gs[l], bes[l], scale, shift);
    }

    hipMemsetAsync(pooled, 0, (size_t)GGRAPHS * HDIM * 4, stream);
    hipMemsetAsync(cntbuf, 0, GGRAPHS * 4, stream);
    pool_kernel<<<(unsigned)((poolThreads + 255) / 256), 256, 0, stream>>>(
        hbuf, scale, shift, batch, pooled, cntbuf);
    head_kernel<<<GGRAPHS, HDIM, 0, stream>>>(pooled, cntbuf, clinical, Wc, bc, out);
}
